// EngramModule_78778290144018
// MI455X (gfx1250) — matrix operand order repaired and checked by reading, not compiled, not run on hardware
//
#include <hip/hip_runtime.h>
#include <hip/hip_bf16.h>

typedef __attribute__((ext_vector_type(16))) __bf16 v16bf;
typedef __attribute__((ext_vector_type(8)))  float  v8f;
typedef __attribute__((ext_vector_type(4)))  int    v4i;

// ---------------- problem constants (match reference) ----------------
static constexpr int BATCH  = 4;
static constexpr int LSEQ   = 4096;
static constexpr int ROWS   = BATCH * LSEQ;   // 16384
static constexpr int HIDDEN = 1024;
static constexpr int EDIM   = 64;
static constexpr int NHEADS = 8;
static constexpr int NSLOT  = 16;             // 2 orders * 8 heads
static constexpr int DIMR   = NSLOT * EDIM;   // 1024 retrieval dim (= GEMM K)
static constexpr int NOUT   = 2 * HIDDEN;     // 2048 (k and v fused)
static constexpr int VOCAB  = 32000;

// CDNA5 async global->LDS staging (direct LDS write, tracked by ASYNCcnt)
#if defined(__has_builtin)
#if __has_builtin(__builtin_amdgcn_global_load_async_to_lds_b128)
#define HAVE_ASYNC_LDS 1
#endif
#if __has_builtin(__builtin_amdgcn_s_wait_asynccnt)
#define HAVE_WAIT_ASYNC_BUILTIN 1
#endif
#endif

#ifdef HAVE_ASYNC_LDS
static __device__ __forceinline__ void async_copy_b128(const unsigned short* g,
                                                       unsigned short* l) {
    __builtin_amdgcn_global_load_async_to_lds_b128(
        (__attribute__((address_space(1))) v4i*)g,
        (__attribute__((address_space(3))) v4i*)l, 0, 0);
}
static __device__ __forceinline__ void wait_async0() {
#ifdef HAVE_WAIT_ASYNC_BUILTIN
    __builtin_amdgcn_s_wait_asynccnt(0);
#else
    asm volatile("s_wait_asynccnt 0" ::: "memory");
#endif
}
#endif

// fp32 -> bf16 round-to-nearest-even
static __device__ __forceinline__ unsigned short f2bf(float f) {
    unsigned int u = __float_as_uint(f);
    u += 0x7FFFu + ((u >> 16) & 1u);
    return (unsigned short)(u >> 16);
}

// ---------------- K0: pack [w_k; w_v] -> bf16 [2048, 1024] ----------------
__global__ __launch_bounds__(256) void k_cvt_w(const float* __restrict__ wk,
                                               const float* __restrict__ wv,
                                               unsigned short* __restrict__ wkv) {
    int idx = blockIdx.x * 256 + threadIdx.x;     // 0 .. 2048*1024-1
    int r = idx >> 10;
    int c = idx & 1023;
    float s = (r < HIDDEN) ? wk[r * DIMR + c] : wv[(r - HIDDEN) * DIMR + c];
    wkv[idx] = f2bf(s);
}

// ---------------- K1: hash + table gather -> memory bf16 [16384, 1024] ----------------
__global__ __launch_bounds__(256) void k_gather(const int* __restrict__ token_ids,
                                                const int* __restrict__ projection,
                                                const int* __restrict__ mults2,
                                                const int* __restrict__ mults3,
                                                const int* __restrict__ table_sizes,
                                                const float* __restrict__ tables,
                                                unsigned short* __restrict__ memA,
                                                int max_s) {
    const int row = blockIdx.x;           // (b, l) flattened
    const int b = row / LSEQ;
    const int l = row % LSEQ;
    const int t = threadIdx.x;

    __shared__ int csh[3];                // compressed[l], [l-1], [l-2] (0 if OOB)
    if (t < 3) {
        int ll = l - t;
        int c = 0;
        if (ll >= 0) {
            int tok = token_ids[b * LSEQ + ll];
            tok = tok < 0 ? 0 : (tok > VOCAB - 1 ? VOCAB - 1 : tok);
            c = projection[tok];
        }
        csh[t] = c;
    }
    __syncthreads();

    const int slot = t >> 4;              // 0..15 : order*8 + head
    const int l16  = t & 15;
    const int order = slot >> 3;
    const int h = slot & 7;

    long long mix;
    if (order == 0) {
        mix = (long long)csh[1] * (long long)mults2[h * 2 + 0]
            ^ (long long)csh[0] * (long long)mults2[h * 2 + 1];
    } else {
        mix = (long long)csh[2] * (long long)mults3[h * 3 + 0]
            ^ (long long)csh[1] * (long long)mults3[h * 3 + 1]
            ^ (long long)csh[0] * (long long)mults3[h * 3 + 2];
    }
    if (mix < 0) mix = -mix;
    long long sz = (long long)table_sizes[order * NHEADS + h];
    long long idx = mix % sz;

    const float* src = tables + ((size_t)slot * (size_t)max_s + (size_t)idx) * EDIM + l16 * 4;
    float4 f = *(const float4*)src;
    ushort4 o;
    o.x = f2bf(f.x); o.y = f2bf(f.y); o.z = f2bf(f.z); o.w = f2bf(f.w);
    *(ushort4*)(memA + (size_t)row * DIMR + slot * EDIM + l16 * 4) = o;
}

// ---------------- K2: WMMA GEMM  kv[16384,2048] = memA[16384,1024] * wkv^T ----------------
// Block tile 128(M) x 128(N), K-step 32, double-buffered LDS with async global->LDS staging.
// 8 waves in 2x4 grid, each wave 64x32 (8 wmma accumulators).
static constexpr int BM  = 128;
static constexpr int BN  = 128;
static constexpr int BK  = 32;
static constexpr int LDT = 40;   // LDS row stride (bf16 elems), 80 B: keeps 16B alignment

__global__ __launch_bounds__(256) void k_gemm(const unsigned short* __restrict__ A,
                                              const unsigned short* __restrict__ Bw,
                                              float* __restrict__ C) {
    __shared__ unsigned short As[2][BM * LDT];
    __shared__ unsigned short Bs[2][BN * LDT];

    const int tid  = threadIdx.x;
    const int lane = tid & 31;
    const int wave = tid >> 5;
    const int wm = wave >> 2;           // 0..1
    const int wn = wave & 3;            // 0..3
    const int half = lane >> 4;         // 0/1
    const int l16  = lane & 15;

    const int gm0 = blockIdx.x * BM;
    const int gn0 = blockIdx.y * BN;

    v8f acc[4][2];
    const v8f vzero = {0.f, 0.f, 0.f, 0.f, 0.f, 0.f, 0.f, 0.f};
#pragma unroll
    for (int i = 0; i < 4; ++i)
#pragma unroll
        for (int j = 0; j < 2; ++j) acc[i][j] = vzero;

    const int srow = tid >> 1;          // 0..127
    const int sseg = (tid & 1) * 16;    // bf16 elem offset 0 / 16

    // per-thread staging addresses (K offset added per step)
    const unsigned short* gA = A  + (size_t)(gm0 + srow) * DIMR + sseg;
    const unsigned short* gB = Bw + (size_t)(gn0 + srow) * DIMR + sseg;
    const int lofs = srow * LDT + sseg;

    constexpr int NSTEP = DIMR / BK;    // 32
    union Frag { uint4 u[2]; v16bf v; };

#ifdef HAVE_ASYNC_LDS
    // ---- prologue: async-stage tile 0 into buffer 0 ----
    async_copy_b128(gA,     &As[0][lofs]);
    async_copy_b128(gA + 8, &As[0][lofs + 8]);
    async_copy_b128(gB,     &Bs[0][lofs]);
    async_copy_b128(gB + 8, &Bs[0][lofs + 8]);
    wait_async0();
    __syncthreads();
#else
    { // sync prologue
        const uint4* pa = (const uint4*)gA;
        const uint4* pb = (const uint4*)gB;
        uint4* la = (uint4*)&As[0][lofs];
        uint4* lb = (uint4*)&Bs[0][lofs];
        la[0] = pa[0]; la[1] = pa[1];
        lb[0] = pb[0]; lb[1] = pb[1];
        __syncthreads();
    }
#endif

    for (int step = 0; step < NSTEP; ++step) {
        const int cur = step & 1;
        const int nxt = cur ^ 1;
        const int knext = (step + 1) * BK;

#ifdef HAVE_ASYNC_LDS
        if (step + 1 < NSTEP) {   // async-stage next tile while computing current
            async_copy_b128(gA + knext,     &As[nxt][lofs]);
            async_copy_b128(gA + knext + 8, &As[nxt][lofs + 8]);
            async_copy_b128(gB + knext,     &Bs[nxt][lofs]);
            async_copy_b128(gB + knext + 8, &Bs[nxt][lofs + 8]);
        }
#else
        uint4 ra0, ra1, rb0, rb1;
        if (step + 1 < NSTEP) {   // prefetch next tile into registers
            const uint4* pa = (const uint4*)(gA + knext);
            const uint4* pb = (const uint4*)(gB + knext);
            ra0 = pa[0]; ra1 = pa[1];
            rb0 = pb[0]; rb1 = pb[1];
        }
#endif

        // ---- compute from buffer `cur` ----
        // B fragments: lane holds column n = l16, k = half*16 + [0..15] (contiguous 32B)
        Frag bf[2];
#pragma unroll
        for (int ni = 0; ni < 2; ++ni) {
            const int n = wn * 32 + ni * 16 + l16;
            const uint4* p = (const uint4*)(&Bs[cur][n * LDT + half * 16]);
            bf[ni].u[0] = p[0];
            bf[ni].u[1] = p[1];
        }
        // A fragments: lane holds row r = l16, k = half*16 + [0..15] (contiguous 32B)
#pragma unroll
        for (int mi = 0; mi < 4; ++mi) {
            const int r = wm * 64 + mi * 16 + l16;
            Frag af;
            af.u[0] = *(const uint4*)(&As[cur][r * LDT + half * 16]);
            af.u[1] = *(const uint4*)(&As[cur][r * LDT + half * 16 + 8]);
#pragma unroll
            for (int ni = 0; ni < 2; ++ni) {
                acc[mi][ni] = __builtin_amdgcn_wmma_f32_16x16x32_bf16(
                    false, af.v, false, bf[ni].v, (short)0, acc[mi][ni], false, false);
            }
        }

#ifdef HAVE_ASYNC_LDS
        if (step + 1 < NSTEP) {
            wait_async0();       // next buffer fully landed in LDS (this wave)
            __syncthreads();     // all waves: landed + done reading `cur`
        }
#else
        if (step + 1 < NSTEP) {
            uint4* la = (uint4*)&As[nxt][lofs];
            uint4* lb = (uint4*)&Bs[nxt][lofs];
            la[0] = ra0; la[1] = ra1;
            lb[0] = rb0; lb[1] = rb1;
            __syncthreads();
        }
#endif
    }

    // epilogue: D layout — lane col n = l16, rows m = half*8 + v
#pragma unroll
    for (int mi = 0; mi < 4; ++mi)
#pragma unroll
        for (int ni = 0; ni < 2; ++ni) {
            const int grow0 = gm0 + wm * 64 + mi * 16 + half * 8;
            const int gcol  = gn0 + wn * 32 + ni * 16 + l16;
            float* cp = C + (size_t)grow0 * NOUT + gcol;
#pragma unroll
            for (int v = 0; v < 8; ++v) cp[(size_t)v * NOUT] = acc[mi][ni][v];
        }
}

// ---------------- K3: per-row fuse (rmsnorm q/k, gate, v_g, rmsnorm v_g) ----------------
// In-place: kv[row, 0:1024] <- h, kv[row, 1024:2048] <- v_g
static __device__ __forceinline__ float blockSum(float v, float* s) {
    const int t = threadIdx.x;
    s[t] = v;
    __syncthreads();
    for (int off = 128; off > 0; off >>= 1) {
        if (t < off) s[t] += s[t + off];
        __syncthreads();
    }
    float r = s[0];
    __syncthreads();
    return r;
}

__global__ __launch_bounds__(256) void k_fuse(const float* __restrict__ hs,
                                              float* __restrict__ kv,
                                              const float* __restrict__ qw,
                                              const float* __restrict__ kw,
                                              const float* __restrict__ cw) {
    __shared__ float sred[256];
    const int row = blockIdx.x;
    const int d = threadIdx.x * 4;

    const float* hrow = hs + (size_t)row * HIDDEN;
    float* kvrow = kv + (size_t)row * NOUT;

    float4 q = *(const float4*)(hrow + d);
    float4 k = *(const float4*)(kvrow + d);
    float4 v = *(const float4*)(kvrow + HIDDEN + d);

    float ssq = blockSum(q.x * q.x + q.y * q.y + q.z * q.z + q.w * q.w, sred);
    float ssk = blockSum(k.x * k.x + k.y * k.y + k.z * k.z + k.w * k.w, sred);
    const float rq = rsqrtf(ssq * (1.0f / HIDDEN) + 1e-6f);
    const float rk = rsqrtf(ssk * (1.0f / HIDDEN) + 1e-6f);

    float4 qwv = *(const float4*)(qw + d);
    float4 kwv = *(const float4*)(kw + d);
    float4 qn, kn;
    qn.x = q.x * rq * qwv.x; qn.y = q.y * rq * qwv.y; qn.z = q.z * rq * qwv.z; qn.w = q.w * rq * qwv.w;
    kn.x = k.x * rk * kwv.x; kn.y = k.y * rk * kwv.y; kn.z = k.z * rk * kwv.z; kn.w = k.w * rk * kwv.w;

    float dp = blockSum(qn.x * kn.x + qn.y * kn.y + qn.z * kn.z + qn.w * kn.w, sred);
    const float gate = 1.0f / (1.0f + expf(-dp * (1.0f / 32.0f)));   // 1/sqrt(1024)

    float4 vg;
    vg.x = gate * v.x; vg.y = gate * v.y; vg.z = gate * v.z; vg.w = gate * v.w;

    float ssv = blockSum(vg.x * vg.x + vg.y * vg.y + vg.z * vg.z + vg.w * vg.w, sred);
    const float rv = rsqrtf(ssv * (1.0f / HIDDEN) + 1e-6f);

    float4 cwv = *(const float4*)(cw + d);
    float4 h;
    h.x = vg.x * rv * cwv.x; h.y = vg.y * rv * cwv.y; h.z = vg.z * rv * cwv.z; h.w = vg.w * rv * cwv.w;

    *(float4*)(kvrow + d) = h;                 // overwrite k-slot with h
    *(float4*)(kvrow + HIDDEN + d) = vg;       // overwrite v-slot with v_g
}

// ---------------- K4: causal dilated depthwise conv (K=4, dil=3) + SiLU + residual ----------------
__global__ __launch_bounds__(256) void k_conv(const float* __restrict__ kv,   // h in [:,0:1024], v_g in [:,1024:2048]
                                              const float* __restrict__ conv_w,  // [1024, 4]
                                              const float* __restrict__ conv_b,  // [1024]
                                              float* __restrict__ out) {
    const int g  = blockIdx.x * 256 + threadIdx.x;   // group of 4 channels
    const int cg = g & 255;
    const int l  = (g >> 8) & (LSEQ - 1);
    const int b  = g >> 20;                          // g / (256*4096)
    const int c4 = cg * 4;

    float4 y;
    y.x = conv_b[c4 + 0]; y.y = conv_b[c4 + 1]; y.z = conv_b[c4 + 2]; y.w = conv_b[c4 + 3];

    float4 w0 = *(const float4*)(conv_w + (c4 + 0) * 4);
    float4 w1 = *(const float4*)(conv_w + (c4 + 1) * 4);
    float4 w2 = *(const float4*)(conv_w + (c4 + 2) * 4);
    float4 w3 = *(const float4*)(conv_w + (c4 + 3) * 4);

#pragma unroll
    for (int j = 0; j < 4; ++j) {
        const int pos = l - 9 + 3 * j;
        if (pos >= 0) {
            float4 hv = *(const float4*)(kv + (size_t)(b * LSEQ + pos) * NOUT + c4);
            const float wj0 = (&w0.x)[j], wj1 = (&w1.x)[j], wj2 = (&w2.x)[j], wj3 = (&w3.x)[j];
            y.x += hv.x * wj0; y.y += hv.y * wj1; y.z += hv.z * wj2; y.w += hv.w * wj3;
        }
    }

    float4 vg = *(const float4*)(kv + (size_t)(b * LSEQ + l) * NOUT + HIDDEN + c4);
    float4 o;
    o.x = y.x / (1.0f + expf(-y.x)) + vg.x;
    o.y = y.y / (1.0f + expf(-y.y)) + vg.y;
    o.z = y.z / (1.0f + expf(-y.z)) + vg.z;
    o.w = y.w / (1.0f + expf(-y.w)) + vg.w;
    *(float4*)(out + (size_t)(b * LSEQ + l) * HIDDEN + c4) = o;
}

// ---------------- host launch ----------------
extern "C" void kernel_launch(void* const* d_in, const int* in_sizes, int n_in,
                              void* d_out, int out_size, void* d_ws, size_t ws_size,
                              hipStream_t stream) {
    const float* hs         = (const float*)d_in[0];
    const int*   token_ids  = (const int*)d_in[1];
    const int*   projection = (const int*)d_in[2];
    const int*   mults2     = (const int*)d_in[3];
    const int*   mults3     = (const int*)d_in[4];
    const int*   tsizes     = (const int*)d_in[5];
    const float* tables     = (const float*)d_in[6];
    const float* w_k        = (const float*)d_in[7];
    const float* w_v        = (const float*)d_in[8];
    const float* q_norm_w   = (const float*)d_in[9];
    const float* k_norm_w   = (const float*)d_in[10];
    const float* conv_norm  = (const float*)d_in[11];
    const float* conv_w     = (const float*)d_in[12];
    const float* conv_b     = (const float*)d_in[13];
    float* out = (float*)d_out;

    const int max_s = in_sizes[6] / (2 * NHEADS * EDIM);

    // workspace layout
    char* ws = (char*)d_ws;
    unsigned short* memA = (unsigned short*)ws;                       // 16384*1024 bf16 = 32 MB
    unsigned short* wkv  = (unsigned short*)(ws + (size_t)ROWS * DIMR * 2);          // 4 MB
    float* kv = (float*)(ws + (size_t)ROWS * DIMR * 2 + (size_t)NOUT * DIMR * 2);    // 128 MB

    // K0: weight convert  (2048*1024 / 256 = 8192 blocks)
    k_cvt_w<<<(NOUT * DIMR) / 256, 256, 0, stream>>>(w_k, w_v, wkv);

    // K1: hash + gather  (one block per token row)
    k_gather<<<ROWS, 256, 0, stream>>>(token_ids, projection, mults2, mults3,
                                       tsizes, tables, memA, max_s);

    // K2: WMMA GEMM  (grid 128 x 16 tiles of 128x128)
    dim3 ggrid(ROWS / BM, NOUT / BN);
    k_gemm<<<ggrid, 256, 0, stream>>>(memA, wkv, kv);

    // K3: per-row fuse
    k_fuse<<<ROWS, 256, 0, stream>>>(hs, kv, q_norm_w, k_norm_w, conv_norm);

    // K4: conv + silu + residual  (B*L*256 groups)
    k_conv<<<(BATCH * LSEQ * 256) / 256, 256, 0, stream>>>(kv, conv_w, conv_b, out);
}
